// DCC_12120397709850
// MI455X (gfx1250) — matrix operand order repaired and checked by reading, not compiled, not run on hardware
//
#include <hip/hip_runtime.h>
#include <math.h>

// ---------------------------------------------------------------------------
// feature [B=128, N=64, D=256] f32 -> rows X [M=8192, D=256].
// loss = mean_r( log(sum_c exp(<x_r,x_c>)) - log(sum_{c in batch(r)} exp) ).
// GEMM is 2*8192^2*256 = 34.4 GFLOP over 4 MB of f16 data (L2-resident on a
// 192 MB L2) -> pure WMMA-bound. Each wave holds TWO 16-row A tiles in
// registers and issues 2 WMMAs per B load (1:1 wmma:load).
// __launch_bounds__(256,1) relaxes the occupancy floor so the ~220 VGPRs of
// A/acc/sums stay in registers (round 2 spilled A to scratch at the default
// 8-waves/SIMD register budget).
// ---------------------------------------------------------------------------
#define B_    128
#define N_    64
#define D_    256
#define M_    (B_ * N_)        // 8192 rows
#define RT_   (M_ / 16)        // 512 16-row/col tiles
#define NWAVE 8                // waves per workgroup (256 threads, wave32)
#define MT    2                // row tiles per workgroup (A register reuse)
#define NWG   (M_ / (MT * 16)) // 256 workgroups for the main kernel

typedef _Float16 v8h  __attribute__((ext_vector_type(8)));
typedef _Float16 v16h __attribute__((ext_vector_type(16)));
typedef float    v8f  __attribute__((ext_vector_type(8)));

// ---------------------------------------------------------------------------
// Kernel 1: L2-normalize each row (f32 math, eps=1e-12), store f16 row-major.
// One wave32 per row, 8 elements per lane.
// ---------------------------------------------------------------------------
__global__ void k_normalize(const float* __restrict__ feat,
                            _Float16* __restrict__ xh) {
  const int wave = threadIdx.x >> 5;
  const int lane = threadIdx.x & 31;
  const int row  = blockIdx.x * NWAVE + wave;

  const float* src = feat + (size_t)row * D_ + lane * 8;
  float v[8];
#pragma unroll
  for (int i = 0; i < 8; ++i) v[i] = src[i];

  float ss = 0.0f;
#pragma unroll
  for (int i = 0; i < 8; ++i) ss += v[i] * v[i];
#pragma unroll
  for (int m = 16; m >= 1; m >>= 1) ss += __shfl_xor(ss, m, 32);

  const float scale = 1.0f / fmaxf(sqrtf(ss), 1e-12f);

  v8h h;
#pragma unroll
  for (int i = 0; i < 8; ++i) h[i] = (_Float16)(v[i] * scale);
  *(v8h*)(xh + (size_t)row * D_ + lane * 8) = h;
}

// ---------------------------------------------------------------------------
// Kernel 2: workgroup owns MT=2 adjacent 16-row tiles (32 rows, all inside
// one 64-row batch block). Waves split the 512 column tiles. Per column tile:
// 8 k-steps x (1 B load -> 2 WMMAs). exp + row-sum accumulation in registers
// (C/D layout: VGPR g <-> row g or g+8, lane <-> column), then lane-half
// butterfly + LDS cross-wave reduction -> per-WG partial loss.
//
// A-layout (16-bit A 16x32): lane l (M=l&15):
//   l<16 : K kb+0..15 in h[0..15]
//   l>=16: K kb+16..31 in h[0..15]
// B-layout (32x16): lane l = column l&15, contiguous K kb+(l>>4)*16..+16.
// ---------------------------------------------------------------------------
__global__ void __launch_bounds__(256, 1)
k_gram_loss(const _Float16* __restrict__ xh,
            float* __restrict__ partial) {
  __shared__ float s_tot[NWAVE][MT][16];
  __shared__ float s_pos[NWAVE][MT][16];
  __shared__ float s_loss[MT * 16];

  const int wave = threadIdx.x >> 5;       // 0..7
  const int lane = threadIdx.x & 31;
  const int r0   = blockIdx.x * (MT * 16); // first row of this WG

  // ---- preload MT A tiles (each 16 rows x 256 K) into 128 VGPRs ----
  const int mlane = lane & 15;
  const int ksel  = (lane >> 4) << 4;      // 0 or 16 (f16 elements)
  v16h Areg[MT][8];
#pragma unroll
  for (int t = 0; t < MT; ++t) {
#pragma unroll
    for (int kt = 0; kt < 8; ++kt) {
      const _Float16* base = xh + (size_t)(r0 + t * 16 + mlane) * D_ + kt * 32 + ksel;
      v8h lo = *(const v8h*)(base);
      v8h hi = *(const v8h*)(base + 8);
      Areg[t][kt] = __builtin_shufflevector(lo, hi,
                      0,1,2,3,4,5,6,7,8,9,10,11,12,13,14,15);
    }
  }

  float tot[MT][8], pos[MT][8];
#pragma unroll
  for (int t = 0; t < MT; ++t)
#pragma unroll
    for (int g = 0; g < 8; ++g) { tot[t][g] = 0.0f; pos[t][g] = 0.0f; }

  const int ncol  = lane & 15;
  const int koff  = (lane >> 4) << 4;      // 0 or 16 (f16 elements)
  const int posHi = r0 >> 6;               // batch block of all 32 rows

  for (int ct = wave; ct < RT_; ct += NWAVE) {
    const _Float16* bbase = xh + (size_t)(ct * 16 + ncol) * D_ + koff;
    v8f acc[MT];
#pragma unroll
    for (int t = 0; t < MT; ++t) acc[t] = (v8f){};
#pragma unroll
    for (int kt = 0; kt < 8; ++kt) {
      v16h b = *(const v16h*)(bbase + kt * 32);   // one 32B load ...
#pragma unroll
      for (int t = 0; t < MT; ++t) {              // ... feeds MT WMMAs
        acc[t] = __builtin_amdgcn_wmma_f32_16x16x32_f16(
                   false, Areg[t][kt], false, b, (short)0, acc[t], false, false);
      }
    }
    const bool isPos = ((ct >> 2) == posHi);
#pragma unroll
    for (int t = 0; t < MT; ++t) {
#pragma unroll
      for (int g = 0; g < 8; ++g) {
        const float e = __expf(acc[t][g]);        // |logit| <= 1, safe
        tot[t][g] += e;
        if (isPos) pos[t][g] += e;
      }
    }
  }

  // reduce across the 16-lane column halves (masks 1,2,4,8 stay in-half)
#pragma unroll
  for (int m = 8; m >= 1; m >>= 1) {
#pragma unroll
    for (int t = 0; t < MT; ++t)
#pragma unroll
      for (int g = 0; g < 8; ++g) {
        tot[t][g] += __shfl_xor(tot[t][g], m, 32);
        pos[t][g] += __shfl_xor(pos[t][g], m, 32);
      }
  }

  // lane 0 holds rows 0..7 (per tile), lane 16 holds rows 8..15
  if ((lane & 15) == 0) {
    const int rbase = (lane >> 4) << 3;
#pragma unroll
    for (int t = 0; t < MT; ++t)
#pragma unroll
      for (int g = 0; g < 8; ++g) {
        s_tot[wave][t][rbase + g] = tot[t][g];
        s_pos[wave][t][rbase + g] = pos[t][g];
      }
  }
  __syncthreads();

  if (threadIdx.x < MT * 16) {
    const int t = threadIdx.x >> 4;
    const int r = threadIdx.x & 15;
    float T = 0.0f, P = 0.0f;
#pragma unroll
    for (int w = 0; w < NWAVE; ++w) { T += s_tot[w][t][r]; P += s_pos[w][t][r]; }
    s_loss[threadIdx.x] = __logf(T) - __logf(P);   // -log(pos/total)
  }
  __syncthreads();

  if (threadIdx.x == 0) {
    float a = 0.0f;
#pragma unroll
    for (int r = 0; r < MT * 16; ++r) a += s_loss[r];
    partial[blockIdx.x] = a;
  }
}

// ---------------------------------------------------------------------------
// Kernel 3: reduce NWG partials -> mean loss scalar.
// ---------------------------------------------------------------------------
__global__ void k_reduce(const float* __restrict__ partial,
                         float* __restrict__ out) {
  __shared__ float s[256];
  float v = 0.0f;
  for (int i = threadIdx.x; i < NWG; i += 256) v += partial[i];
  s[threadIdx.x] = v;
  __syncthreads();
  for (int stride = 128; stride >= 1; stride >>= 1) {
    if (threadIdx.x < stride) s[threadIdx.x] += s[threadIdx.x + stride];
    __syncthreads();
  }
  if (threadIdx.x == 0) out[0] = s[0] * (1.0f / (float)M_);
}

// ---------------------------------------------------------------------------
// Launch: normalize -> fused WMMA gram/softmax-sums -> final reduce.
// Workspace: [0, 4MB) f16 X; then NWG floats of tile partials.
// ---------------------------------------------------------------------------
extern "C" void kernel_launch(void* const* d_in, const int* in_sizes, int n_in,
                              void* d_out, int out_size, void* d_ws, size_t ws_size,
                              hipStream_t stream) {
  const float* feat = (const float*)d_in[0];
  _Float16* xh      = (_Float16*)d_ws;
  float* partial    = (float*)((char*)d_ws + (size_t)M_ * D_ * sizeof(_Float16));
  float* out        = (float*)d_out;

  k_normalize<<<M_ / NWAVE, 256, 0, stream>>>(feat, xh);
  k_gram_loss<<<NWG, 256, 0, stream>>>(xh, partial);
  k_reduce<<<1, 256, 0, stream>>>(partial, out);
}
